// MaskSRU_24137716204149
// MI455X (gfx1250) — matrix operand order repaired and checked by reading, not compiled, not run on hardware
//
#include <hip/hip_runtime.h>
#include <stdint.h>

// Problem constants (reference: B=16, T=1024, D=1024, H=1024)
#define B_    16
#define T_    1024
#define D_    1024
#define H_    1024
#define N3H   3072
#define MTOT  (B_ * T_)   // 16384 rows

typedef __attribute__((ext_vector_type(16))) __bf16 v16bf;
typedef __attribute__((ext_vector_type(16))) short  v16s;
typedef __attribute__((ext_vector_type(8)))  short  v8s;
typedef __attribute__((ext_vector_type(8)))  float  v8f;

static __device__ __forceinline__ unsigned short f32_to_bf16_rne(float f) {
    union { float f; uint32_t u; } v; v.f = f;
    uint32_t u = v.u;
    uint32_t r = u + 0x7FFFu + ((u >> 16) & 1u);   // round-to-nearest-even
    if ((u & 0x7F800000u) == 0x7F800000u) r = u;   // keep inf/nan payload
    return (unsigned short)(r >> 16);
}

// ---------------------------------------------------------------------------
// Kernel 1: mask input by seq_lens and convert fp32 -> bf16 (row-major [M,K])
// ---------------------------------------------------------------------------
__global__ __launch_bounds__(256) void sru_mask_cvt_x(
    const float* __restrict__ x, const int* __restrict__ seq_lens,
    unsigned short* __restrict__ xb) {
    int idx4 = blockIdx.x * blockDim.x + threadIdx.x;  // float4 index
    int row  = idx4 >> 8;          // D/4 = 256 float4 per row
    int b    = row >> 10;          // row / T
    int t    = row & 1023;         // row % T
    float4 v = ((const float4*)x)[idx4];
    float  m = (t < seq_lens[b]) ? 1.0f : 0.0f;
    ushort4 o;
    o.x = f32_to_bf16_rne(v.x * m);
    o.y = f32_to_bf16_rne(v.y * m);
    o.z = f32_to_bf16_rne(v.z * m);
    o.w = f32_to_bf16_rne(v.w * m);
    ((ushort4*)xb)[idx4] = o;
}

// ---------------------------------------------------------------------------
// Kernel 2: LDS-tiled transpose + convert: W[K=1024, N=3072] fp32
//           -> Wt[N=3072, K=1024] bf16.  Block (32,8), 32x32 tiles.
// ---------------------------------------------------------------------------
__global__ __launch_bounds__(256) void sru_cvt_w_t(
    const float* __restrict__ w, unsigned short* __restrict__ wt) {
    __shared__ float tile[32][33];
    const int tx = threadIdx.x;            // 0..31
    const int ty = threadIdx.y;            // 0..7
    const int n0 = blockIdx.x * 32;
    const int k0 = blockIdx.y * 32;
    #pragma unroll
    for (int j = 0; j < 4; ++j) {
        int kk = ty + j * 8;
        tile[kk][tx] = w[(size_t)(k0 + kk) * N3H + n0 + tx];   // coalesced in n
    }
    __syncthreads();
    #pragma unroll
    for (int j = 0; j < 4; ++j) {
        int nn = ty + j * 8;
        wt[(size_t)(n0 + nn) * D_ + k0 + tx] =                 // coalesced in k
            f32_to_bf16_rne(tile[tx][nn]);
    }
}

// ---------------------------------------------------------------------------
// Kernel 3: bf16 WMMA GEMM.  U[dir] = Xbf16 @ W[dir]   (B supplied as Wt[N,K])
//   M = 16384, N = 3072, K = 1024.  Block tile 128x256, BK = 64.
//   8 waves (2x4); each wave owns 64x64 = 4x4 WMMA 16x16 subtiles.
//   Both LDS tiles are K-contiguous, so every fragment is 2x ds_load_b128.
// ---------------------------------------------------------------------------
#define BM  128
#define BN  256
#define BK  64
#define LDK 72    // 144 B/row: 16B aligned, bank stride 36 (conflict-free)

__global__ __launch_bounds__(256) void sru_gemm_bf16(
    const unsigned short* __restrict__ XB,    // [M, K] bf16
    const unsigned short* __restrict__ WtF,   // [N, K] bf16
    const unsigned short* __restrict__ WtB,   // [N, K] bf16
    float* __restrict__ UF, float* __restrict__ UB) {
    __shared__ unsigned short As[BM * LDK];   // 128 rows (M) x 64 (K)
    __shared__ unsigned short Bs[BN * LDK];   // 256 rows (N) x 64 (K)

    const unsigned short* __restrict__ Wt = (blockIdx.z == 0) ? WtF : WtB;
    float* __restrict__ U = (blockIdx.z == 0) ? UF : UB;

    const int tid  = threadIdx.x;
    const int lane = tid & 31;          // wave32
    const int wid  = tid >> 5;          // 0..7
    const int wm   = (wid >> 2) * 64;   // wave row offset: 0 or 64
    const int wn   = (wid & 3)  * 64;   // wave col offset: 0..192
    const int m0   = blockIdx.x * BM;
    const int n0   = blockIdx.y * BN;

    const int lm   = lane & 15;         // sub-row / sub-col within 16
    const int kbA  = (lane >> 4) * 8;   // A: hi lanes hold K+8..15 / K+24..31
    const int kbB  = (lane >> 4) * 16;  // unused; B fragments are loaded with kbA, same K order as A

    // staging maps (256 threads)
    const int aRow = tid >> 1;          // 0..127 (M)
    const int aCol = (tid & 1) * 32;    // 0 or 32 (K)
    const int bRow = tid >> 2;          // 0..63  (N), +64*rb
    const int bCol = (tid & 3) * 16;    // 0..48  (K)

    v8f acc[4][4];
    #pragma unroll
    for (int i = 0; i < 4; ++i)
        #pragma unroll
        for (int j = 0; j < 4; ++j) { v8f z = {}; acc[i][j] = z; }

    for (int k0 = 0; k0 < D_; k0 += BK) {
        // ---- stage A (128x64) and B (256x64) into LDS, 16B vector moves ----
        const unsigned short* ga = XB + (size_t)(m0 + aRow) * D_ + k0 + aCol;
        unsigned short* la = As + aRow * LDK + aCol;
        #pragma unroll
        for (int q = 0; q < 4; ++q)
            *(v8s*)(la + 8 * q) = *(const v8s*)(ga + 8 * q);
        #pragma unroll
        for (int rb = 0; rb < 4; ++rb) {
            const int r = bRow + rb * 64;
            const unsigned short* gb = Wt + (size_t)(n0 + r) * D_ + k0 + bCol;
            unsigned short* lb = Bs + r * LDK + bCol;
            *(v8s*)(lb)     = *(const v8s*)(gb);
            *(v8s*)(lb + 8) = *(const v8s*)(gb + 8);
        }
        if (k0 + BK < D_) {
            __builtin_prefetch(ga + BK, 0, 1);                      // next A
            __builtin_prefetch(Wt + (size_t)(n0 + bRow) * D_ + k0 + BK, 0, 1);
        }
        __syncthreads();

        // ---- two K-substeps of 32 ----
        #pragma unroll
        for (int ks = 0; ks < 2; ++ks) {
            const int koff = ks * 32;
            v16bf afrag[4];
            #pragma unroll
            for (int sm = 0; sm < 4; ++sm) {
                const unsigned short* Ar = As + (wm + sm * 16 + lm) * LDK + koff;
                v8s lo = *(const v8s*)(Ar + kbA);        // K koff+kbA..+7
                v8s hi = *(const v8s*)(Ar + 16 + kbA);   // K koff+16+kbA..+7
                v16s av = __builtin_shufflevector(lo, hi,
                    0, 1, 2, 3, 4, 5, 6, 7, 8, 9, 10, 11, 12, 13, 14, 15);
                afrag[sm] = __builtin_bit_cast(v16bf, av);
            }
            v16bf bfrag[4];
            #pragma unroll
            for (int sn = 0; sn < 4; ++sn) {
                const unsigned short* Br = Bs + (wn + sn * 16 + lm) * LDK + koff;
                v8s lo = *(const v8s*)(Br + kbA);        // K koff+kbA..+7
                v8s hi = *(const v8s*)(Br + 16 + kbA);   // K koff+16+kbA..+7
                v16s bv = __builtin_shufflevector(lo, hi,
                    0, 1, 2, 3, 4, 5, 6, 7, 8, 9, 10, 11, 12, 13, 14, 15);
                bfrag[sn] = __builtin_bit_cast(v16bf, bv);
            }
            #pragma unroll
            for (int sm = 0; sm < 4; ++sm)
                #pragma unroll
                for (int sn = 0; sn < 4; ++sn)
                    acc[sm][sn] = __builtin_amdgcn_wmma_f32_16x16x32_bf16(
                        false, afrag[sm], false, bfrag[sn],
                        (short)0, acc[sm][sn], false, false);
        }
        __syncthreads();
    }

    // ---- epilogue: C/D layout (VGPR r -> M=r / M=r+8, lane&15 -> N) ----
    #pragma unroll
    for (int sm = 0; sm < 4; ++sm) {
        #pragma unroll
        for (int sn = 0; sn < 4; ++sn) {
            const int outCol = n0 + wn + sn * 16 + lm;
            #pragma unroll
            for (int r = 0; r < 8; ++r) {
                int mrow = m0 + wm + sm * 16 + ((lane < 16) ? r : r + 8);
                U[(size_t)mrow * N3H + outCol] = acc[sm][sn][r];
            }
        }
    }
}

// ---------------------------------------------------------------------------
// Kernel 4: fused SRU scan + highway + mask + concat.
//   One thread per (b,h) channel; blockIdx.y = direction (0 fwd, 1 bwd).
// ---------------------------------------------------------------------------
__global__ __launch_bounds__(256) void sru_scan(
    const float* __restrict__ x, const int* __restrict__ seq_lens,
    const float* __restrict__ UF, const float* __restrict__ UB,
    const float* __restrict__ b_fwd, const float* __restrict__ b_bwd,
    float* __restrict__ out) {
    const int g   = blockIdx.x * blockDim.x + threadIdx.x;  // 0..B*H-1
    const int dir = blockIdx.y;
    const int b   = g >> 10;     // g / H
    const int h   = g & 1023;    // g % H
    const float* __restrict__ U    = dir ? UB : UF;
    const float* __restrict__ bias = dir ? b_bwd : b_fwd;
    const float bf = bias[h];
    const float br = bias[H_ + h];
    const int   L  = seq_lens[b];

    float c = 0.0f;
    for (int s = 0; s < T_; ++s) {
        const int t = dir ? (T_ - 1 - s) : s;
        const size_t row = (size_t)(b * T_ + t);
        const float* Up = U + row * N3H;
        const float xt = Up[h];             // x_tilde
        const float fp = Up[H_ + h];        // forget pre-act
        const float rp = Up[2 * H_ + h];    // reset pre-act
        const float f = 1.0f / (1.0f + __expf(-(fp + bf)));
        const float r = 1.0f / (1.0f + __expf(-(rp + br)));
        c = f * c + (1.0f - f) * xt;
        const float xv = (t < L) ? x[row * (size_t)D_ + h] : 0.0f;  // masked x
        const float hv = r * tanhf(c) + (1.0f - r) * xv;
        out[row * (size_t)(2 * H_) + (size_t)dir * H_ + h] = (t < L) ? hv : 0.0f;
    }
}

// ---------------------------------------------------------------------------
// Host launcher
// ---------------------------------------------------------------------------
extern "C" void kernel_launch(void* const* d_in, const int* in_sizes, int n_in,
                              void* d_out, int out_size, void* d_ws, size_t ws_size,
                              hipStream_t stream) {
    const float* x   = (const float*)d_in[0];   // [B,T,D]
    const int*   seq = (const int*)  d_in[1];   // [B]
    const float* Wf  = (const float*)d_in[2];   // [D,3H]
    const float* bf  = (const float*)d_in[3];   // [2H]
    const float* Wb  = (const float*)d_in[4];   // [D,3H]
    const float* bb  = (const float*)d_in[5];   // [2H]
    float* out = (float*)d_out;                 // [B,T,2H]

    char* ws = (char*)d_ws;
    const size_t szXB = (size_t)MTOT * D_ * sizeof(unsigned short);   // 32 MB
    const size_t szW  = (size_t)D_ * N3H * sizeof(unsigned short);    // 6 MB
    const size_t szU  = (size_t)MTOT * N3H * sizeof(float);           // 192 MB
    unsigned short* XB  = (unsigned short*)(ws);
    unsigned short* WtF = (unsigned short*)(ws + szXB);
    unsigned short* WtB = (unsigned short*)(ws + szXB + szW);
    float* UF = (float*)(ws + szXB + 2 * szW);
    float* UB = (float*)(ws + szXB + 2 * szW + szU);

    // 1) mask + convert activations; transpose + convert weights
    sru_mask_cvt_x<<<(MTOT * D_ / 4) / 256, 256, 0, stream>>>(x, seq, XB);
    dim3 wt_grid(N3H / 32, D_ / 32);
    dim3 wt_blk(32, 8);
    sru_cvt_w_t<<<wt_grid, wt_blk, 0, stream>>>(Wf, WtF);
    sru_cvt_w_t<<<wt_grid, wt_blk, 0, stream>>>(Wb, WtB);

    // 2) both direction GEMMs: grid.z selects W/U
    dim3 gg(MTOT / BM, N3H / BN, 2);
    sru_gemm_bf16<<<gg, 256, 0, stream>>>(XB, WtF, WtB, UF, UB);

    // 3) fused scan + highway + mask into concatenated output
    dim3 gs((B_ * H_) / 256, 2);
    sru_scan<<<gs, 256, 0, stream>>>(x, seq, UF, UB, bf, bb, out);
}
